// Qwen3MoeSparseMoeBlock_84473416778092
// MI455X (gfx1250) — matrix operand order repaired and checked by reading, not compiled, not run on hardware
//
#include <hip/hip_runtime.h>

#define T_TOK 2048
#define H_DIM 2048
#define E_NUM 64
#define I_DIM 768
#define TOPK  8
#define CAP   2048     // worst-case tokens per expert
#define TILE_M 16
#define KCHUNK 256

typedef __attribute__((ext_vector_type(16))) __bf16          v16bf;
typedef __attribute__((ext_vector_type(2)))  __bf16          v2bf;
typedef __attribute__((ext_vector_type(16))) unsigned short  v16us;
typedef __attribute__((ext_vector_type(8)))  unsigned int    v8u;
typedef __attribute__((ext_vector_type(8)))  float           v8f;
typedef int v4i __attribute__((vector_size(16)));

#if defined(__gfx1250__) && __has_builtin(__builtin_amdgcn_global_load_async_to_lds_b128)
#define USE_ASYNC 1
#else
#define USE_ASYNC 0
#endif

// Pack two fp32 -> packed bf16x2 dword.
#if defined(__gfx1250__) && __has_builtin(__builtin_amdgcn_cvt_pk_bf16_f32)
// Hardware packed convert (RNE), one VALU op: v_cvt_pk_bf16_f32. Low half = a.
__device__ __forceinline__ unsigned int pack_bf16x2(float a, float b) {
  v2bf v = __builtin_amdgcn_cvt_pk_bf16_f32(a, b);
  return __builtin_bit_cast(unsigned int, v);
}
#else
// Fallback: round-half-up + one v_perm_b32.
__device__ __forceinline__ unsigned int pack_bf16x2(float a, float b) {
  unsigned int ua = __float_as_uint(a) + 0x8000u;
  unsigned int ub = __float_as_uint(b) + 0x8000u;
  // result bytes: [1:0] = ua[3:2], [3:2] = ub[3:2]
  return __builtin_amdgcn_perm(ub, ua, 0x07060302u);
}
#endif

__device__ __forceinline__ unsigned short f32_bf16(float f) {
  return (unsigned short)((__float_as_uint(f) + 0x8000u) >> 16);
}

// A-fragment (16x32 bf16, MxK) from fp32 LDS tile, row-major [16][stride] floats.
// Lanes 0-15: M=lane, elems 0..7 -> K=kb..kb+7, elems 8..15 -> K=kb+16..kb+23.
// Lanes 16-31: M=lane-16, K offsets +8 / +24.
__device__ __forceinline__ v16bf frag_a_lds_f32(const float* base, int stride,
                                                int lane, int kbase) {
  int m  = lane & 15;
  int o0 = kbase + ((lane & 16) ? 8 : 0);
  const float* p0 = base + m * stride + o0;
  const float* p1 = p0 + 16;
  v8u r;
#pragma unroll
  for (int i = 0; i < 4; ++i) r[i]     = pack_bf16x2(p0[2 * i], p0[2 * i + 1]);
#pragma unroll
  for (int i = 0; i < 4; ++i) r[4 + i] = pack_bf16x2(p1[2 * i], p1[2 * i + 1]);
  return __builtin_bit_cast(v16bf, r);
}

// A-fragment from a bf16 (ushort) LDS tile, row-major [16][stride].
__device__ __forceinline__ v16bf frag_a_lds_bf16(const unsigned short* base, int stride,
                                                 int lane, int kbase) {
  int m  = lane & 15;
  int o0 = kbase + ((lane & 16) ? 8 : 0);
  const unsigned short* p0 = base + m * stride + o0;
  const unsigned short* p1 = p0 + 16;
  v16us r;
#pragma unroll
  for (int i = 0; i < 8; ++i) { r[i] = p0[i]; r[i + 8] = p1[i]; }
  return __builtin_bit_cast(v16bf, r);
}

// B-fragment (32x16 bf16, KxN) from global fp32 weights, row-major [N][rstride],
// fp32 -> bf16 in-flight.
// Lanes 0-15: N=lane, K=kb..kb+7, kb+16..kb+23. Lanes 16-31: N=lane-16, K offsets +8 / +24.
__device__ __forceinline__ v16bf frag_b_global(const float* __restrict__ w,
                                               long long rstride, int lane, int kbase) {
  int n  = lane & 15;
  int kb = kbase + ((lane & 16) ? 8 : 0);
  const float* p = w + (long long)n * rstride + kb;
  v8u r;
#pragma unroll
  for (int i = 0; i < 8; ++i) r[i] = pack_bf16x2(p[2 * i + (i >> 2) * 8], p[2 * i + 1 + (i >> 2) * 8]);
  return __builtin_bit_cast(v16bf, r);
}

__global__ void zero_kernel(float* __restrict__ out, int n, int* __restrict__ counts) {
  int i = blockIdx.x * blockDim.x + threadIdx.x;
  for (; i < n; i += gridDim.x * blockDim.x) out[i] = 0.f;
  if (blockIdx.x == 0 && threadIdx.x < E_NUM) counts[threadIdx.x] = 0;
}

// One block per token, 64 threads (one per expert logit).
__global__ void router_kernel(const float* __restrict__ x, const float* __restrict__ gw,
                              int* __restrict__ counts, int* __restrict__ toks,
                              float* __restrict__ wts) {
  __shared__ float sx[H_DIM];
  __shared__ float sl[E_NUM];
  int t = blockIdx.x;
  int tid = threadIdx.x;
  for (int i = tid; i < H_DIM; i += 64) sx[i] = x[(long long)t * H_DIM + i];
  __syncthreads();
  const float* g = gw + (long long)tid * H_DIM;
  float acc = 0.f;
  for (int k = 0; k < H_DIM; ++k) acc = fmaf(sx[k], g[k], acc);
  sl[tid] = acc;
  __syncthreads();
  if (tid == 0) {
    float mx = -1e30f;
    for (int e = 0; e < E_NUM; ++e) mx = fmaxf(mx, sl[e]);
    float p[E_NUM];
    float sum = 0.f;
    for (int e = 0; e < E_NUM; ++e) { p[e] = __expf(sl[e] - mx); sum += p[e]; }
    float inv = 1.f / sum;
    int   idx[TOPK];
    float val[TOPK];
    float vsum = 0.f;
    for (int k = 0; k < TOPK; ++k) {
      int bi = 0; float bv = -1.f;
      for (int e = 0; e < E_NUM; ++e) if (p[e] > bv) { bv = p[e]; bi = e; }
      p[bi] = -2.f;
      idx[k] = bi;
      val[k] = bv * inv;
      vsum  += val[k];
    }
    float rinv = 1.f / vsum;                       // top-k renormalization
    for (int k = 0; k < TOPK; ++k) {
      int e = idx[k];
      int pos = atomicAdd(&counts[e], 1);
      if (pos < CAP) {
        toks[e * CAP + pos] = t;
        wts[e * CAP + pos]  = val[k] * rinv;
      }
    }
  }
}

// One block per (expert, 16-token tile): fused gate/up -> silu*mul -> down -> combine.
__global__ void __launch_bounds__(256)
moe_kernel(const float* __restrict__ x,
           const float* __restrict__ wgu, const float* __restrict__ wd,
           const int* __restrict__ counts, const int* __restrict__ toks,
           const float* __restrict__ wts, float* __restrict__ out) {
  int e    = blockIdx.x >> 7;           // 128 tiles per expert
  int tile = blockIdx.x & 127;
  int cnt  = counts[e];
  if (cnt > CAP) cnt = CAP;
  int base = tile * TILE_M;
  if (base >= cnt) return;              // uniform early-exit (before any barrier)
  int rows = min(TILE_M, cnt - base);

  __shared__ float          sXf[TILE_M * KCHUNK];  // 16 KB fp32 activation chunk
  __shared__ unsigned short sH[TILE_M * I_DIM];    // 24 KB bf16 intermediate h
  __shared__ int   sTok[TILE_M];
  __shared__ float sW[TILE_M];

  int tid  = threadIdx.x;
  int lane = tid & 31;
  int wave = tid >> 5;

  if (tid < TILE_M) {
    if (tid < rows) {
      sTok[tid] = toks[e * CAP + base + tid];
      sW[tid]   = wts[e * CAP + base + tid];
    } else {
      sTok[tid] = 0;
      sW[tid]   = 0.f;
    }
  }
  // Zero pad rows once; staging below only touches rows < `rows`.
  for (int i = rows * KCHUNK + tid; i < TILE_M * KCHUNK; i += 256) sXf[i] = 0.f;
  __syncthreads();

  const float* wgu_e = wgu + (long long)e * (2 * I_DIM) * H_DIM;
  const float* wd_e  = wd  + (long long)e * H_DIM * I_DIM;

  // ---- Stage A: gu = X @ Wgu^T (bf16 WMMA, f32 acc); 12 accumulators per wave ----
  v8f accG[6], accU[6];
#pragma unroll
  for (int p = 0; p < 6; ++p) { accG[p] = (v8f){}; accU[p] = (v8f){}; }

  for (int kc = 0; kc < H_DIM; kc += KCHUNK) {
#if USE_ASYNC
    // Async DMA: 16 rows x 256 fp32 = 16 KB; each thread moves 16B x 4 issues.
#pragma unroll
    for (int it = 0; it < 4; ++it) {
      int f = (tid + it * 256) * 4;           // flat float index
      int r = f >> 8;                         // 256 floats per row
      int c = f & 255;
      if (r < rows) {
        __builtin_amdgcn_global_load_async_to_lds_b128(
            (__attribute__((address_space(1))) v4i*)
                (x + (long long)sTok[r] * H_DIM + kc + c),
            (__attribute__((address_space(3))) v4i*)(&sXf[f]),
            0, 0);
      }
    }
#if __has_builtin(__builtin_amdgcn_s_wait_asynccnt)
    __builtin_amdgcn_s_wait_asynccnt(0);
#else
    asm volatile("s_wait_asynccnt 0" ::: "memory");
#endif
#else
    for (int i = tid; i < rows * KCHUNK; i += 256) {
      int r = i >> 8, c = i & 255;
      sXf[i] = x[(long long)sTok[r] * H_DIM + kc + c];
    }
#endif
    __syncthreads();
#pragma unroll 1
    for (int k0 = 0; k0 < KCHUNK; k0 += 32) {
      v16bf a = frag_a_lds_f32(sXf, KCHUNK, lane, k0);
#pragma unroll
      for (int p = 0; p < 6; ++p) {
        int c0 = p * 128 + wave * 16;   // gate column tile base (0..768)
        v16bf bg = frag_b_global(wgu_e + (long long)c0 * H_DIM, H_DIM, lane, kc + k0);
        v16bf bu = frag_b_global(wgu_e + (long long)(c0 + I_DIM) * H_DIM, H_DIM, lane, kc + k0);
        accG[p] = __builtin_amdgcn_wmma_f32_16x16x32_bf16(false, a, false, bg,
                                                          (short)0, accG[p], false, false);
        accU[p] = __builtin_amdgcn_wmma_f32_16x16x32_bf16(false, a, false, bu,
                                                          (short)0, accU[p], false, false);
      }
    }
    __syncthreads();
  }

  // SiLU(gate) * up -> sH (bf16). C layout: lanes 0-15: N=lane, M=r; lanes 16-31: N=lane-16, M=r+8.
  {
    int n     = lane & 15;
    int mbase = (lane & 16) ? 8 : 0;
#pragma unroll
    for (int p = 0; p < 6; ++p) {
      int col = p * 128 + wave * 16 + n;
#pragma unroll
      for (int r = 0; r < 8; ++r) {
        float g = accG[p][r];
        float u = accU[p][r];
        float h = (g / (1.f + __expf(-g))) * u;
        sH[(mbase + r) * I_DIM + col] = f32_bf16(h);
      }
    }
  }
  __syncthreads();

  // ---- Stage B: y = h @ Wd^T, then weighted atomic combine into out ----
#pragma unroll 1
  for (int p = 0; p < 16; ++p) {
    int n0 = p * 128 + wave * 16;       // output column tile base (0..2048)
    if (n0 + 128 < H_DIM)               // speculative prefetch of next weight rows
      __builtin_prefetch(wd_e + (long long)(n0 + 128) * I_DIM + (lane << 4), 0, 1);
    v8f acc = (v8f){};
#pragma unroll 1
    for (int k0 = 0; k0 < I_DIM; k0 += 32) {
      v16bf a = frag_a_lds_bf16(sH, I_DIM, lane, k0);
      v16bf b = frag_b_global(wd_e + (long long)n0 * I_DIM, I_DIM, lane, k0);
      acc = __builtin_amdgcn_wmma_f32_16x16x32_bf16(false, a, false, b,
                                                    (short)0, acc, false, false);
    }
    int n     = lane & 15;
    int mbase = (lane & 16) ? 8 : 0;
#pragma unroll
    for (int r = 0; r < 8; ++r) {
      int m = mbase + r;
      if (m < rows) {
        atomicAdd(&out[(long long)sTok[m] * H_DIM + n0 + n], sW[m] * acc[r]);
      }
    }
  }
}

extern "C" void kernel_launch(void* const* d_in, const int* in_sizes, int n_in,
                              void* d_out, int out_size, void* d_ws, size_t ws_size,
                              hipStream_t stream) {
  const float* x   = (const float*)d_in[0];  // [T, H]
  const float* gw  = (const float*)d_in[1];  // [E, H]
  const float* wgu = (const float*)d_in[2];  // [E, 2I, H]
  const float* wd  = (const float*)d_in[3];  // [E, H, I]
  float* out = (float*)d_out;                // [T, H] fp32

  char*  ws     = (char*)d_ws;
  int*   counts = (int*)ws;                                   // 64 ints
  int*   toks   = (int*)(ws + 256);                           // 64*2048 ints
  float* wts    = (float*)(ws + 256 + (size_t)E_NUM * CAP * 4);

  zero_kernel<<<256, 256, 0, stream>>>(out, T_TOK * H_DIM, counts);
  router_kernel<<<T_TOK, 64, 0, stream>>>(x, gw, counts, toks, wts);
  moe_kernel<<<E_NUM * (T_TOK / TILE_M), 256, 0, stream>>>(x, wgu, wd,
                                                           counts, toks, wts, out);
}
